// EncoderLayer_hybrid2_77747497992482
// MI455X (gfx1250) — matrix operand order repaired and checked by reading, not compiled, not run on hardware
//
#include <hip/hip_runtime.h>
#include <stdint.h>

#define B_  4
#define S_  2048
#define NH_ 8
#define DH_ 32
#define DK_ 256

typedef __attribute__((ext_vector_type(16))) __bf16          v16bf;
typedef __attribute__((ext_vector_type(16))) unsigned short  v16u;
typedef __attribute__((ext_vector_type(8)))  unsigned int    v8ui;
typedef __attribute__((ext_vector_type(16))) float           v16f;
typedef __attribute__((ext_vector_type(8)))  float           v8f;

__device__ __forceinline__ unsigned short f2bf(float f) {
  unsigned int u = __builtin_bit_cast(unsigned int, f);
  u += 0x7FFFu + ((u >> 16) & 1u);            // round-to-nearest-even
  return (unsigned short)(u >> 16);
}

// Pack two floats as bf16 pair (round-to-nearest, ties up): 2 adds + 1 v_perm_b32.
__device__ __forceinline__ unsigned int pack_bf16x2(float a, float b) {
  unsigned int ua = __builtin_bit_cast(unsigned int, a) + 0x8000u;
  unsigned int ub = __builtin_bit_cast(unsigned int, b) + 0x8000u;
#if __has_builtin(__builtin_amdgcn_perm)
  return __builtin_amdgcn_perm(ub, ua, 0x07060302u);  // {ub[31:16], ua[31:16]}
#else
  return (ua >> 16) | (ub & 0xFFFF0000u);
#endif
}

__device__ __forceinline__ float fast_exp2(float x) {
#if __has_builtin(__builtin_amdgcn_exp2f)
  return __builtin_amdgcn_exp2f(x);           // v_exp_f32
#else
  return __expf(x * 0.6931471805599453f);
#endif
}

__device__ __forceinline__ float fast_rcp(float x) {
#if __has_builtin(__builtin_amdgcn_rcpf)
  return __builtin_amdgcn_rcpf(x);            // v_rcp_f32
#else
  return 1.0f / x;
#endif
}

__device__ __forceinline__ float bcast_lane(float v, int l) {
#if __has_builtin(__builtin_amdgcn_readlane)
  return __builtin_bit_cast(float,
           __builtin_amdgcn_readlane(__builtin_bit_cast(int, v), l));
#else
  return __shfl(v, l, 32);
#endif
}

__device__ __forceinline__ float wave_sum32(float v) {
#pragma unroll
  for (int off = 16; off > 0; off >>= 1) v += __shfl_xor(v, off, 32);
  return v;
}

// ---------------------------------------------------------------------------
// Kernel 1: LayerNorm + per-head V projection + scalar Q/K.
// One wave per (b,s).  V is written PRE-TRANSPOSED as bf16 [b,h,d,s] (staged
// through LDS so global stores are 16B chunks) so the attention kernel can
// fetch WMMA B-fragments as two contiguous b128 loads per lane.
// Kb = 2*s*log2e*K,  Kc = -s*log2e*K^2  (exp2-domain logit coefficients).
// ---------------------------------------------------------------------------
__global__ __launch_bounds__(256) void prep_kernel(
    const float* __restrict__ x,  const float* __restrict__ Wv,
    const float* __restrict__ bv, const float* __restrict__ wq,
    const float* __restrict__ bq, const float* __restrict__ wk,
    const float* __restrict__ bk,
    unsigned short* __restrict__ Vt, float* __restrict__ Qs,
    float* __restrict__ Kb, float* __restrict__ Kc) {
  __shared__ unsigned short vstage[2048];          // [h*32+d][8 s-slots]

  const int wave = threadIdx.x >> 5;
  const int lane = threadIdx.x & 31;
  const int pos0 = blockIdx.x * 8;                 // 8 consecutive (b,s), same b
  const int pos  = pos0 + wave;
  const int b    = pos >> 11;
  const int s    = pos & (S_ - 1);
  const int s0   = pos0 & (S_ - 1);

  const float* xp = x + (size_t)pos * DK_;
  float xr[8];
  float s1 = 0.f, s2 = 0.f;
#pragma unroll
  for (int k = 0; k < 8; ++k) {
    float v = xp[k * 32 + lane];
    xr[k] = v; s1 += v; s2 += v * v;
  }
  s1 = wave_sum32(s1); s2 = wave_sum32(s2);
  const float mu   = s1 * (1.f / 256.f);
  const float var  = s2 * (1.f / 256.f) - mu * mu;
  const float rstd = rsqrtf(var + 1e-5f);
#pragma unroll
  for (int k = 0; k < 8; ++k) xr[k] = (xr[k] - mu) * rstd;

  const float scale = 0.17677669529663687f;        // 1/sqrt(DH)
  const float l2e   = 1.4426950408889634f;
#pragma unroll
  for (int h = 0; h < NH_; ++h) {
    const float xv = xr[h];                        // xn[h*32 + lane]
    float acc = bv[h * 32 + lane];
#pragma unroll
    for (int e = 0; e < 32; ++e)
      acc = fmaf(bcast_lane(xv, e), Wv[(h * 32 + e) * 32 + lane], acc);
    vstage[(h * 32 + lane) * 8 + wave] = f2bf(acc);

    const float q  = wave_sum32(xv * wq[h * 32 + lane]);
    const float kk = wave_sum32(xv * wk[h * 32 + lane]);
    if (lane == 0) {
      const float Q = q + bq[h];
      const float K = kk + bk[h];
      const size_t o = ((size_t)b * NH_ + h) * S_ + s;
      Qs[o] = Q;
      Kb[o] = 2.f * scale * l2e * K;
      Kc[o] = -scale * l2e * K * K;
    }
  }
  __syncthreads();
  // copy out: thread t owns (h,d) = (t>>5, t&31); 8 bf16 along s = one uint4
  const int t = threadIdx.x;
  const uint4 v = *(const uint4*)&vstage[t * 8];
  *(uint4*)(Vt + ((size_t)(b * NH_ + (t >> 5)) * DH_ + (t & 31)) * S_ + s0) = v;
}

// ---------------------------------------------------------------------------
// Kernel 2: fused flash-attention (L2 scores) + head-mix + residual.
// grid = (S/16, B); 256 threads = 8 waves; wave w handles head h=w for the
// same 16-row i-tile.  Inner loop has NO LDS and NO barriers:
//   - P tile computed analytically per lane in the WMMA A-fragment layout
//     (base term dropped: softmax shift-invariant), exp2-domain.
//   - B fragments: two aligned b128 global loads from pre-transposed V
//     (4 MB total -> L2 resident).
//   - Softmax denominators via a third WMMA against an all-ones B matrix:
//     acc2[v] is exactly the row-sum needed by the D-fragment lane layout.
// Epilogue: per-head outputs exchanged through LDS, 8x8 head mix + bias +
// residual written directly to d_out.
// ---------------------------------------------------------------------------
__global__ __launch_bounds__(256) void attn_merge_kernel(
    const unsigned short* __restrict__ Vt, const float* __restrict__ Qs,
    const float* __restrict__ Kb, const float* __restrict__ Kc,
    const float* __restrict__ x, const float* __restrict__ Wm,
    const float* __restrict__ bm, float* __restrict__ out) {
  __shared__ float ho[16][NH_][DH_];               // [row][head][d]
  __shared__ float wmL[64];
  __shared__ float bmL[8];

  const int h    = threadIdx.x >> 5;               // wave == head
  const int lane = threadIdx.x & 31;
  const int n    = lane & 15;
  const int hi   = lane >> 4;
  const int b    = blockIdx.y;
  const int i0   = blockIdx.x * 16;
  const int i    = i0 + n;

  if (threadIdx.x < 64) wmL[threadIdx.x] = Wm[threadIdx.x];
  if (threadIdx.x < 8)  bmL[threadIdx.x] = bm[threadIdx.x];

  const int    bh  = b * NH_ + h;
  const size_t bhS = (size_t)bh * S_;
  const float  Qi  = Qs[bhS + i];
  const float* kbp = Kb + bhS + 8 * hi;            // lane's first logit run
  const float* kcp = Kc + bhS + 8 * hi;
  // B-fragment rows for this lane: d = n and d = n+16, K runs start at 8*hi and 16+8*hi
  const unsigned short* v0p = Vt + ((size_t)bh * DH_ + n) * S_ + 8 * hi;
  const unsigned short* v1p = v0p + (size_t)16 * S_;

  v16u ou;
#pragma unroll
  for (int e = 0; e < 16; ++e) ou[e] = 0x3F80;     // bf16 1.0
  const v16bf Bone = __builtin_bit_cast(v16bf, ou);

  struct F4x4 { float4 a, b, c, d; };
  struct U4x2 { uint4 a, b; };

  v8f acc0 = {}; v8f acc1 = {}; v8f acc2 = {};

  for (int j0 = 0; j0 < S_; j0 += 32) {
    // logit coefficients: A-element e maps to j = j0 + (e<8 ? e+8*hi : e+8+8*hi)
    F4x4 kbs, kcs;
    kbs.a = *(const float4*)(kbp + j0);      kbs.b = *(const float4*)(kbp + j0 + 4);
    kbs.c = *(const float4*)(kbp + j0 + 16); kbs.d = *(const float4*)(kbp + j0 + 20);
    kcs.a = *(const float4*)(kcp + j0);      kcs.b = *(const float4*)(kcp + j0 + 4);
    kcs.c = *(const float4*)(kcp + j0 + 16); kcs.d = *(const float4*)(kcp + j0 + 20);
    const v16f kbv = __builtin_bit_cast(v16f, kbs);
    const v16f kcv = __builtin_bit_cast(v16f, kcs);

    U4x2 sv0, sv1;
    sv0.a = *(const uint4*)(v0p + j0); sv0.b = *(const uint4*)(v0p + j0 + 16);
    sv1.a = *(const uint4*)(v1p + j0); sv1.b = *(const uint4*)(v1p + j0 + 16);
    const v16bf B0 = __builtin_bit_cast(v16bf, sv0);
    const v16bf B1 = __builtin_bit_cast(v16bf, sv1);

    v8ui au;
#pragma unroll
    for (int k = 0; k < 8; ++k) {
      const float p0 = fast_exp2(fmaf(Qi, kbv[2 * k],     kcv[2 * k]));
      const float p1 = fast_exp2(fmaf(Qi, kbv[2 * k + 1], kcv[2 * k + 1]));
      au[k] = pack_bf16x2(p0, p1);
    }
    const v16bf A = __builtin_bit_cast(v16bf, au);

    acc0 = __builtin_amdgcn_wmma_f32_16x16x32_bf16(false, A, false, B0,
                                                   (short)0, acc0, false, false);
    acc1 = __builtin_amdgcn_wmma_f32_16x16x32_bf16(false, A, false, B1,
                                                   (short)0, acc1, false, false);
    acc2 = __builtin_amdgcn_wmma_f32_16x16x32_bf16(false, A, false, Bone,
                                                   (short)0, acc2, false, false);
  }

  // D layout: VGPR v -> row m = v + 8*hi, column n; acc2[v] = row-sum(P).
#pragma unroll
  for (int v = 0; v < 8; ++v) {
    const int   m  = 8 * hi + v;
    const float rd = fast_rcp(acc2[v]);
    ho[m][h][n]      = acc0[v] * rd;
    ho[m][h][16 + n] = acc1[v] * rd;
  }
  __syncthreads();

  // head mix + bias + residual: out[b, i0+r, d*8+g] = sum_h ho[r][h][d]*Wm[g,h]+bm[g]+x
  const int t = threadIdx.x;
  const int r = t >> 4, q = t & 15;                // row r, channel group q
  const size_t pos = (size_t)b * S_ + i0 + r;
  float hv0[8], hv1[8];
#pragma unroll
  for (int hh = 0; hh < 8; ++hh) {
    hv0[hh] = ho[r][hh][q * 2];
    hv1[hh] = ho[r][hh][q * 2 + 1];
  }
  const float4* xr4 = (const float4*)(x + pos * DK_ + q * 16);
  float4*       or4 = (float4*)(out + pos * DK_ + q * 16);
#pragma unroll
  for (int kq = 0; kq < 4; ++kq) {
    const float4 xv = xr4[kq];
    float res[4];
#pragma unroll
    for (int j2 = 0; j2 < 4; ++j2) {
      const int k = kq * 4 + j2;                   // channel c = q*16+k
      const int g = k & 7;
      const float* hv = (k >> 3) ? hv1 : hv0;      // d = q*2 + (k>>3)
      float acc = bmL[g];
#pragma unroll
      for (int hh = 0; hh < 8; ++hh) acc = fmaf(hv[hh], wmL[g * 8 + hh], acc);
      res[j2] = acc;
    }
    float4 o;
    o.x = res[0] + xv.x; o.y = res[1] + xv.y;
    o.z = res[2] + xv.z; o.w = res[3] + xv.w;
    or4[kq] = o;
  }
}

// ---------------------------------------------------------------------------
extern "C" void kernel_launch(void* const* d_in, const int* in_sizes, int n_in,
                              void* d_out, int out_size, void* d_ws, size_t ws_size,
                              hipStream_t stream) {
  (void)in_sizes; (void)n_in; (void)out_size; (void)ws_size;
  const float* x  = (const float*)d_in[0];
  const float* Wv = (const float*)d_in[1];
  const float* bv = (const float*)d_in[2];
  const float* wq = (const float*)d_in[3];
  const float* bq = (const float*)d_in[4];
  const float* wk = (const float*)d_in[5];
  const float* bk = (const float*)d_in[6];
  const float* Wm = (const float*)d_in[7];
  const float* bm = (const float*)d_in[8];
  float* out = (float*)d_out;

  char* ws = (char*)d_ws;
  unsigned short* Vt = (unsigned short*)ws;               // [B,NH,DH,S] bf16: 4 MB
  float* Qs = (float*)(ws + (4u << 20));                  // [B,NH,S] f32
  float* Kb = Qs + (size_t)B_ * NH_ * S_;
  float* Kc = Kb + (size_t)B_ * NH_ * S_;

  prep_kernel<<<(B_ * S_) / 8, 256, 0, stream>>>(x, Wv, bv, wq, bq, wk, bk,
                                                 Vt, Qs, Kb, Kc);
  attn_merge_kernel<<<dim3(S_ / 16, B_), 256, 0, stream>>>(Vt, Qs, Kb, Kc,
                                                           x, Wm, bm, out);
}
